// Dynamic_Visit_77747497992701
// MI455X (gfx1250) — compile-verified, matrix operand order checked by reading, not run on hardware
//
#include <hip/hip_runtime.h>
#include <hip/hip_bf16.h>

typedef __attribute__((ext_vector_type(16))) _Float16 v16h;
typedef __attribute__((ext_vector_type(8)))  float    v8f;

#define N_PTS     512
#define N_LIGHTS  100
#define N_RAYS    (N_PTS * N_LIGHTS)
#define N_SAMPLES 50
#define N_IMP     30
#define IN_FINE   80
#define HID       256
#define NEARV     0.1f
#define FARV      3.0f

// workspace layout (bytes)
#define WT_ELEMS  507904                              // total f16 elems of transposed weights
#define ALPHA_OFF ((size_t)WT_ELEMS * 2)              // 1,015,808
#define COS_OFF   (ALPHA_OFF + (size_t)N_RAYS * IN_FINE * 2)  // + 8,192,000

struct WPtrs { const float* w[8]; };
struct BPtrs { const float* b[8]; const float* w_last; const float* b_last; };

__device__ __forceinline__ float eval_sigma(float x, float y, float z,
                                            const float* __restrict__ wf, float bf) {
  // NeRF positional encoding: [x, {sin(2^f x), cos(2^f x)}_{f=0..3}] . wf + bf
  float s = bf + wf[0] * x + wf[1] * y + wf[2] * z;
  float fr = 1.0f;
  int base = 3;
#pragma unroll
  for (int f = 0; f < 4; ++f) {
    s += wf[base + 0] * __sinf(x * fr) + wf[base + 1] * __sinf(y * fr) + wf[base + 2] * __sinf(z * fr);
    s += wf[base + 3] * __cosf(x * fr) + wf[base + 4] * __cosf(y * fr) + wf[base + 5] * __cosf(z * fr);
    base += 6;
    fr *= 2.0f;
  }
  return s;
}

// ---------------------------------------------------------------------------
// Kernel 0: convert fp32 weights [din][256] -> f16 column-major [256][Kpad]
// so WMMA B-fragment K-pairs are contiguous dwords. Zero-fill K padding.
// ---------------------------------------------------------------------------
__global__ __launch_bounds__(256) void prep_weights_kernel(WPtrs wp, _Float16* __restrict__ wT) {
  const int Kpad[8] = {96, 256, 256, 256, 256, 352, 256, 256};
  const int din[8]  = {80, 256, 256, 256, 256, 336, 256, 256};
  const int off[8]  = {0, 24576, 90112, 155648, 221184, 286720, 376832, 442368};
  int l = blockIdx.y;
  int idx = blockIdx.x * 256 + threadIdx.x;
  int kp = Kpad[l];
  if (idx >= 256 * kp) return;
  int n = idx / kp;
  int k = idx - n * kp;
  float v = (k < din[l]) ? wp.w[l][k * HID + n] : 0.0f;
  wT[off[l] + idx] = (_Float16)v;
}

// ---------------------------------------------------------------------------
// Kernel 1: per-ray coarse pass, importance sampling, fine alphas.
// One thread per ray. Writes alpha_f (f16) and cosin (f32) to workspace.
// ---------------------------------------------------------------------------
__global__ __launch_bounds__(256) void ray_prep_kernel(
    const float* __restrict__ pts, const float* __restrict__ lights,
    const float* __restrict__ sin_colat, const float* __restrict__ normal,
    const float* __restrict__ rand_bias, const float* __restrict__ wf,
    const float* __restrict__ bf_p,
    _Float16* __restrict__ alphaB, float* __restrict__ cosB)
{
  int g = blockIdx.x * blockDim.x + threadIdx.x;
  if (g >= N_RAYS) return;
  int p  = g / N_LIGHTS;
  int li = g - p * N_LIGHTS;

  float px = pts[p * 3 + 0], py = pts[p * 3 + 1], pz = pts[p * 3 + 2];
  float dx = lights[li * 3 + 0] - px;
  float dy = lights[li * 3 + 1] - py;
  float dzc = lights[li * 3 + 2] - pz;
  float inv = rsqrtf(dx * dx + dy * dy + dzc * dzc);
  dx *= inv; dy *= inv; dzc *= inv;

  float nx = normal[p * 3 + 0], ny = normal[p * 3 + 1], nz = normal[p * 3 + 2];
  cosB[g] = (dx * nx + dy * ny + dzc * nz > 0.0f) ? 1.0f : 0.0f;

  float sc = sin_colat[li];
  float bf = bf_p[0];
  const float dzv = (FARV - NEARV) / (float)(N_SAMPLES - 1);

  // coarse: weights = alpha * transmittance
  float w[N_SAMPLES];
  float T = 1.0f;
  for (int i = 0; i < N_SAMPLES; ++i) {
    float z  = NEARV + dzv * (float)i;
    float rb = rand_bias[(size_t)g * N_SAMPLES + i];
    float bb = sc * z * 0.01f * (rb * 2.0f - 1.0f);
    float sg = eval_sigma(px + dx * z + bb, py + dy * z + bb, pz + dzc * z + bb, wf, bf);
    float dist = (i == N_SAMPLES - 1) ? 1e10f : dzv;
    float a = 1.0f - __expf(-fmaxf(sg, 0.0f) * dist);
    w[i] = a * T;
    T *= (1.0f - a + 1e-10f);
  }

  // CDF over weights[1..48] (48 bins), cdf has 49 entries with cdf[0]=0
  float cdf[N_SAMPLES - 1];
  cdf[0] = 0.0f;
  float tot = 0.0f;
  for (int j = 1; j <= 48; ++j) tot += w[j] + 1e-5f;
  float run = 0.0f, itot = 1.0f / tot;
  for (int j = 1; j <= 48; ++j) { run += w[j] + 1e-5f; cdf[j] = run * itot; }

  // deterministic inverse-CDF samples over bins z_mid[j] = NEAR + dzv*(j+0.5)
  float zs[N_IMP];
  int ind = 0;
  for (int j = 0; j < N_IMP; ++j) {
    float u = (float)j / (float)(N_IMP - 1);
    while (ind < 49 && cdf[ind] <= u) ++ind;   // searchsorted(side='right'), u ascending
    int below = ind - 1; if (below < 0) below = 0;
    int above = ind;     if (above > 48) above = 48;
    float c0 = cdf[below], c1 = cdf[above];
    float b0 = NEARV + dzv * ((float)below + 0.5f);
    float b1 = NEARV + dzv * ((float)above + 0.5f);
    float den = c1 - c0; if (den < 1e-5f) den = 1.0f;
    float t = (u - c0) / den;
    zs[j] = b0 + t * (b1 - b0);
  }

  // merge two sorted lists -> z_all[80]
  float zall[IN_FINE];
  int i50 = 0, i30 = 0;
  for (int k = 0; k < IN_FINE; ++k) {
    float zv = NEARV + dzv * (float)i50;
    if (i30 >= N_IMP || (i50 < N_SAMPLES && zv <= zs[i30])) { zall[k] = zv; ++i50; }
    else { zall[k] = zs[i30]; ++i30; }
  }

  // fine alphas
  for (int k = 0; k < IN_FINE; ++k) {
    float dist = (k == IN_FINE - 1) ? 1e10f : (zall[k + 1] - zall[k]);
    float z = zall[k];
    float sg = eval_sigma(px + dx * z, py + dy * z, pz + dzc * z, wf, bf);
    float a = 1.0f - __expf(-fmaxf(sg, 0.0f) * dist);
    alphaB[(size_t)g * IN_FINE + k] = (_Float16)a;
  }
}

// ---------------------------------------------------------------------------
// Kernel 2: 8-layer MLP via v_wmma_f32_16x16x32_f16.
// 128 threads (4 waves) per block, 64 rays per block.
// LDS [64][352] f16: cols 0..79 = alpha (kept for skip concat),
// cols 80..335 = current hidden, cols 336..351 = zero K-pad.
// ---------------------------------------------------------------------------
#define ROWS 64
#define KBUF 352

__global__ __launch_bounds__(128) void mlp_wmma_kernel(
    const _Float16* __restrict__ alphaB, const float* __restrict__ cosB,
    const _Float16* __restrict__ wT, BPtrs bp, float* __restrict__ out)
{
  __shared__ _Float16 hbuf[ROWS][KBUF];

  const int kStart[8] = {0, 80, 80, 80, 80, 0, 80, 80};
  const int kTiles[8] = {3, 8, 8, 8, 8, 11, 8, 8};
  const int Kpad[8]   = {96, 256, 256, 256, 256, 352, 256, 256};
  const int wOff[8]   = {0, 24576, 90112, 155648, 221184, 286720, 376832, 442368};

  int tid  = threadIdx.x;
  int wave = tid >> 5;
  int lane = tid & 31;
  int hh   = lane >> 4;     // lane half: selects K sub-block per ISA layout
  int l16  = lane & 15;
  int m0   = wave * 16;     // this wave's 16-row tile
  int rayBase = blockIdx.x * ROWS;

  // zero-init LDS (establishes K padding), then load alpha into cols 0..79
  _Float16* hflat = &hbuf[0][0];
  for (int i = tid; i < ROWS * KBUF; i += 128) hflat[i] = (_Float16)0.0f;
  __syncthreads();
  for (int i = tid; i < ROWS * IN_FINE; i += 128) {
    int r = i / IN_FINE, c = i - r * IN_FINE;
    hbuf[r][c] = alphaB[(size_t)(rayBase + r) * IN_FINE + c];
  }
  __syncthreads();

  for (int l = 0; l < 8; ++l) {
    v8f acc[16];
    v8f vz = {};
#pragma unroll
    for (int nt = 0; nt < 16; ++nt) acc[nt] = vz;

    const int ks  = kStart[l];
    const int ktn = kTiles[l];
    const int kp  = Kpad[l];
    const _Float16* wl = wT + wOff[l];

    for (int kt = 0; kt < ktn; ++kt) {
      // A fragment: 16x32 f16 per ISA layout. lane<16: K 0-7 (v0-3) & 16-23 (v4-7);
      // lane>=16: K 8-15 & 24-31. Row M = m0 + (lane&15). K pairs contiguous in LDS.
      union { v16h v; unsigned int u[8]; } A;
      int kb  = kt * 32;
      int row = m0 + l16;
#pragma unroll
      for (int r = 0; r < 8; ++r) {
        int k = kb + ((r < 4) ? (hh * 8 + 2 * r) : (16 + hh * 8 + 2 * (r - 4)));
        A.u[r] = *(const unsigned int*)&hbuf[row][ks + k];
      }
#pragma unroll
      for (int nt = 0; nt < 16; ++nt) {
        // B fragment: 32x16, mirror layout with N = nt*16 + (lane&15);
        // weights are col-major [N][Kpad] so K pairs are contiguous dwords.
        union { v16h v; unsigned int u[8]; } Bf;
        int n = nt * 16 + l16;
        const _Float16* col = wl + (size_t)n * kp;
#pragma unroll
        for (int r = 0; r < 8; ++r) {
          int k = kb + ((r < 4) ? (hh * 8 + 2 * r) : (16 + hh * 8 + 2 * (r - 4)));
          Bf.u[r] = *(const unsigned int*)&col[k];
        }
        acc[nt] = __builtin_amdgcn_wmma_f32_16x16x32_f16(
            false, A.v, false, Bf.v, (short)0, acc[nt], false, false);
      }
    }

    __syncthreads();  // all waves done reading hbuf for this layer
    const float* bias = bp.b[l];
#pragma unroll
    for (int nt = 0; nt < 16; ++nt) {
      int n = nt * 16 + l16;
      float bv = bias[n];
#pragma unroll
      for (int r = 0; r < 8; ++r) {
        // C layout: lane holds (M = r + half*8, N = lane&15) in VGPR r
        float v = acc[nt][r] + bv;
        v = v > 0.0f ? v : 0.0f;          // ReLU
        hbuf[m0 + hh * 8 + r][IN_FINE + n] = (_Float16)v;
      }
    }
    __syncthreads();
  }

  // head: sigmoid(h @ w_last + b_last) * cosin
  if (tid < ROWS) {
    int ray = rayBase + tid;
    float s = bp.b_last[0];
    const float* wl = bp.w_last;
    for (int k = 0; k < HID; ++k) s += (float)hbuf[tid][IN_FINE + k] * wl[k];
    float sig = 1.0f / (1.0f + __expf(-s));
    out[ray] = sig * cosB[ray];
  }
}

// ---------------------------------------------------------------------------
extern "C" void kernel_launch(void* const* d_in, const int* in_sizes, int n_in,
                              void* d_out, int out_size, void* d_ws, size_t ws_size,
                              hipStream_t stream) {
  (void)in_sizes; (void)n_in; (void)out_size; (void)ws_size;

  const float* pts       = (const float*)d_in[0];
  const float* lights    = (const float*)d_in[1];
  const float* sin_colat = (const float*)d_in[2];
  const float* normal    = (const float*)d_in[3];
  const float* rand_bias = (const float*)d_in[4];
  const float* wf        = (const float*)d_in[5];
  const float* bf        = (const float*)d_in[6];

  WPtrs wp; BPtrs bp;
  for (int i = 0; i < 8; ++i) {
    wp.w[i] = (const float*)d_in[9 + 2 * i];
    bp.b[i] = (const float*)d_in[10 + 2 * i];
  }
  bp.w_last = (const float*)d_in[7];
  bp.b_last = (const float*)d_in[8];

  char* ws = (char*)d_ws;
  _Float16* wT     = (_Float16*)ws;
  _Float16* alphaB = (_Float16*)(ws + ALPHA_OFF);
  float*    cosB   = (float*)(ws + COS_OFF);

  dim3 gw(352, 8);
  prep_weights_kernel<<<gw, 256, 0, stream>>>(wp, wT);
  ray_prep_kernel<<<(N_RAYS + 255) / 256, 256, 0, stream>>>(
      pts, lights, sin_colat, normal, rand_bias, wf, bf, alphaB, cosB);
  mlp_wmma_kernel<<<N_RAYS / ROWS, 128, 0, stream>>>(alphaB, cosB, wT, bp, (float*)d_out);
}
